// ResidualVectorQuantizer_40089224740883
// MI455X (gfx1250) — matrix operand order repaired and checked by reading, not compiled, not run on hardware
//
#include <hip/hip_runtime.h>
#include <hip/hip_bf16.h>

typedef __attribute__((ext_vector_type(16))) __bf16 v16bf;
typedef __attribute__((ext_vector_type(8)))  __bf16 v8bf;
typedef __attribute__((ext_vector_type(8)))  float  v8f;

#define RVQ_B 4096
#define RVQ_C 16384
#define RVQ_D 512
#define RVQ_K 8

// ---------------------------------------------------------------------------
// A-fragment (bf16 source): 16x32, rows rowBase..+15, k in [k0, k0+32)
// ISA layout: lanes 0-15 hold K = {k0+0..7, k0+16..23}; lanes 16-31 hold
//             K = {k0+8..15, k0+24..31}; M = lane & 15.
// Each half is 8 contiguous bf16 = 16 bytes = one global_load_b128.
// ---------------------------------------------------------------------------
__device__ inline v16bf load_a_frag(const __bf16* __restrict__ R,
                                    int rowBase, int k0, int lane) {
  int m    = lane & 15;
  int kSel = (lane >> 4) << 3;                 // 0 or 8
  const __bf16* p = R + (long)(rowBase + m) * RVQ_D + k0 + kSel;
  v8bf lo = *(const v8bf*)(p);
  v8bf hi = *(const v8bf*)(p + 16);
  return __builtin_shufflevector(lo, hi, 0, 1, 2, 3, 4, 5, 6, 7,
                                         8, 9, 10, 11, 12, 13, 14, 15);
}

// ---------------------------------------------------------------------------
// B-fragment (bf16 source): 32x16 with B[k][n] = CB[colBase+n][k].
// ISA layout: lanes 0-15 hold K = {k0+0..7, k0+16..23}; lanes 16-31 hold
//             K = {k0+8..15, k0+24..31}; N = lane & 15.
// Each half is 8 contiguous bf16 = 16 bytes = one global_load_b128.
// ---------------------------------------------------------------------------
__device__ inline v16bf load_b_frag(const __bf16* __restrict__ CB,
                                    int colBase, int k0, int lane) {
  int n    = lane & 15;
  int kSel = (lane >> 4) << 3;                 // 0 or 8
  const __bf16* p = CB + (long)(colBase + n) * RVQ_D + k0 + kSel;
  v8bf lo = *(const v8bf*)(p);
  v8bf hi = *(const v8bf*)(p + 16);
  return __builtin_shufflevector(lo, hi, 0, 1, 2, 3, 4, 5, 6, 7,
                                         8, 9, 10, 11, 12, 13, 14, 15);
}

// ---------------------------------------------------------------------------
// Init: residual = z (f32 + bf16 copies), quantized = 0, loss = 0
// ---------------------------------------------------------------------------
__global__ __launch_bounds__(256) void rvq_init(const float* __restrict__ z,
                                                float* __restrict__ residual,
                                                __bf16* __restrict__ residualBf,
                                                float* __restrict__ quant,
                                                float* __restrict__ lossOut) {
  long i = (long)blockIdx.x * blockDim.x + threadIdx.x;
  if (i < (long)RVQ_B * RVQ_D) {
    float v = z[i];
    residual[i]   = v;
    residualBf[i] = (__bf16)v;
    quant[i]      = 0.0f;
  }
  if (i == 0) *lossOut = 0.0f;
}

// ---------------------------------------------------------------------------
// f32 -> bf16 bulk conversion (one codebook slice per step; 8 elems/thread).
// ---------------------------------------------------------------------------
__global__ __launch_bounds__(256) void rvq_cvt(const float* __restrict__ src,
                                               __bf16* __restrict__ dst) {
  long i = ((long)blockIdx.x * blockDim.x + threadIdx.x) * 8;
  const float4* s = (const float4*)(src + i);
  float4 a = s[0], b = s[1];
  v8bf o;
  o[0] = (__bf16)a.x; o[1] = (__bf16)a.y; o[2] = (__bf16)a.z; o[3] = (__bf16)a.w;
  o[4] = (__bf16)b.x; o[5] = (__bf16)b.y; o[6] = (__bf16)b.z; o[7] = (__bf16)b.w;
  *(v8bf*)(dst + i) = o;
}

// ---------------------------------------------------------------------------
// Codebook squared norms: one wave per codebook row, all 8 codebooks at once.
// ---------------------------------------------------------------------------
__global__ __launch_bounds__(256) void rvq_cnorm(const float* __restrict__ cbs,
                                                 float* __restrict__ cnorms) {
  int  wave = threadIdx.x >> 5;
  int  lane = threadIdx.x & 31;
  long row  = (long)blockIdx.x * 8 + wave;     // [0, 8*16384)
  const float* p = cbs + row * RVQ_D;
  float s = 0.0f;
#pragma unroll 4
  for (int i = lane; i < RVQ_D; i += 32) {
    float v = p[i];
    s += v * v;
  }
#pragma unroll
  for (int off = 16; off >= 1; off >>= 1) s += __shfl_xor(s, off, 32);
  if (lane == 0) cnorms[row] = s;
}

// ---------------------------------------------------------------------------
// Fused bf16-WMMA GEMM + argmin. One WG = 16 rows x all 16384 columns.
// 8 waves own interleaved 16-column tiles; K=512 = 16 WMMAs of 16x16x32 bf16.
// 4 rotating accumulators put every WMMA's C-input 4 issues back (covers the
// 1+4-slot 16-bit WMMA hazard window); B loads are left inline so the
// compiler's scheduler pipelines them (clause pairs, waits >= 2 in flight).
// Epilogue tracks running min(‖c‖² − 2·r·c) per row, shfl/LDS reduces.
// ---------------------------------------------------------------------------
__global__ __launch_bounds__(256) void rvq_argmin(const __bf16* __restrict__ R,
                                                  const __bf16* __restrict__ CB,
                                                  const float* __restrict__ cnorm,
                                                  int* __restrict__ bestIdx) {
  const int lane    = threadIdx.x & 31;
  const int wave    = threadIdx.x >> 5;
  const int rowBase = blockIdx.x * 16;

  // Register-resident A fragments for the whole column sweep (16 x v16bf).
  v16bf aFrag[16];
#pragma unroll
  for (int kt = 0; kt < 16; ++kt)
    aFrag[kt] = load_a_frag(R, rowBase, kt * 32, lane);

  float bestD[8];
  int   bestC[8];
#pragma unroll
  for (int r = 0; r < 8; ++r) { bestD[r] = __builtin_inff(); bestC[r] = 0; }

  const int n16 = lane & 15;
  for (int tile = 0; tile < RVQ_C / (16 * 8); ++tile) {
    const int c0 = (tile * 8 + wave) * 16;

    v8f acc0 = {}, acc1 = {}, acc2 = {}, acc3 = {};
#pragma unroll
    for (int kt = 0; kt < 16; kt += 4) {
      v16bf b0 = load_b_frag(CB, c0, (kt + 0) * 32, lane);
      v16bf b1 = load_b_frag(CB, c0, (kt + 1) * 32, lane);
      v16bf b2 = load_b_frag(CB, c0, (kt + 2) * 32, lane);
      v16bf b3 = load_b_frag(CB, c0, (kt + 3) * 32, lane);
      acc0 = __builtin_amdgcn_wmma_f32_16x16x32_bf16(
          false, aFrag[kt + 0], false, b0, (short)0, acc0, false, false);
      acc1 = __builtin_amdgcn_wmma_f32_16x16x32_bf16(
          false, aFrag[kt + 1], false, b1, (short)0, acc1, false, false);
      acc2 = __builtin_amdgcn_wmma_f32_16x16x32_bf16(
          false, aFrag[kt + 2], false, b2, (short)0, acc2, false, false);
      acc3 = __builtin_amdgcn_wmma_f32_16x16x32_bf16(
          false, aFrag[kt + 3], false, b3, (short)0, acc3, false, false);
    }

    const float cn = cnorm[c0 + n16];
    const int   c  = c0 + n16;
#pragma unroll
    for (int r = 0; r < 8; ++r) {
      float dot = (acc0[r] + acc1[r]) + (acc2[r] + acc3[r]);
      float d   = __builtin_fmaf(-2.0f, dot, cn);
      if (d < bestD[r]) { bestD[r] = d; bestC[r] = c; }
    }
  }

  // Reduce across the 16 lanes of each half-wave (column dimension).
#pragma unroll
  for (int off = 8; off >= 1; off >>= 1) {
#pragma unroll
    for (int r = 0; r < 8; ++r) {
      float od = __shfl_xor(bestD[r], off, 16);
      int   oc = __shfl_xor(bestC[r], off, 16);
      if (od < bestD[r] || (od == bestD[r] && oc < bestC[r])) {
        bestD[r] = od; bestC[r] = oc;
      }
    }
  }

  __shared__ float sD[8][16];
  __shared__ int   sC[8][16];
  if ((lane & 15) == 0) {
    int half = lane >> 4;                       // 0: rows 0-7, 1: rows 8-15
#pragma unroll
    for (int r = 0; r < 8; ++r) {
      sD[wave][half * 8 + r] = bestD[r];
      sC[wave][half * 8 + r] = bestC[r];
    }
  }
  __syncthreads();

  if (threadIdx.x < 16) {
    int   row = threadIdx.x;
    float bd  = sD[0][row];
    int   bc  = sC[0][row];
#pragma unroll
    for (int w = 1; w < 8; ++w) {
      float od = sD[w][row];
      int   oc = sC[w][row];
      if (od < bd || (od == bd && oc < bc)) { bd = od; bc = oc; }
    }
    bestIdx[rowBase + row] = bc;
  }
}

// ---------------------------------------------------------------------------
// Per-row update: gather q = cb[idx], quant += q, residual -= q (f32 + bf16),
// loss += 1.25 * mean((q - z)^2), write index (as float).
// ---------------------------------------------------------------------------
__global__ __launch_bounds__(128) void rvq_update(const float* __restrict__ z,
                                                  const float* __restrict__ CB,
                                                  const int* __restrict__ bestIdx,
                                                  float* __restrict__ residual,
                                                  __bf16* __restrict__ residualBf,
                                                  float* __restrict__ quant,
                                                  float* __restrict__ outIdx,
                                                  float* __restrict__ lossOut,
                                                  int step) {
  const int  row = blockIdx.x;
  const int  idx = bestIdx[row];
  const float* q = CB + (long)idx * RVQ_D;
  float s = 0.0f;
#pragma unroll 4
  for (int t = threadIdx.x; t < RVQ_D; t += 128) {
    long  o  = (long)row * RVQ_D + t;
    float qv = q[t];
    quant[o] += qv;
    float rv = residual[o] - qv;
    residual[o]   = rv;
    residualBf[o] = (__bf16)rv;
    float dlt = qv - z[o];
    s += dlt * dlt;
  }
  int wave = threadIdx.x >> 5, lane = threadIdx.x & 31;
#pragma unroll
  for (int off = 16; off >= 1; off >>= 1) s += __shfl_xor(s, off, 32);
  __shared__ float sw[4];
  if (lane == 0) sw[wave] = s;
  __syncthreads();
  if (threadIdx.x == 0) {
    float tot = sw[0] + sw[1] + sw[2] + sw[3];
    atomicAdd(lossOut, 1.25f * tot / (float)((long)RVQ_B * RVQ_D));
    outIdx[(long)row * RVQ_K + step] = (float)idx;
  }
}

// ---------------------------------------------------------------------------
extern "C" void kernel_launch(void* const* d_in, const int* in_sizes, int n_in,
                              void* d_out, int out_size, void* d_ws, size_t ws_size,
                              hipStream_t stream) {
  const float* z   = (const float*)d_in[0];                 // [4096, 512]
  const float* cbs = (const float*)d_in[1];                 // [8, 16384, 512]

  float* out     = (float*)d_out;
  float* quant   = out;                                     // [4096*512]
  float* lossOut = out + (long)RVQ_B * RVQ_D;               // [1]
  float* outIdx  = lossOut + 1;                             // [4096*8]

  // Workspace layout (~30 MB total):
  char*   ws         = (char*)d_ws;
  float*  residual   = (float*)ws;                          // 8 MB
  __bf16* residualBf = (__bf16*)(ws + 8388608L);            // 4 MB
  __bf16* cbBf       = (__bf16*)(ws + 12582912L);           // 16 MB (1 codebook)
  int*    bestIdx    = (int*)(ws + 29360128L);              // 16 KB
  float*  cnorms     = (float*)(ws + 29376512L);            // 512 KB

  rvq_init<<<((long)RVQ_B * RVQ_D + 255) / 256, 256, 0, stream>>>(
      z, residual, residualBf, quant, lossOut);
  rvq_cnorm<<<RVQ_K * RVQ_C / 8, 256, 0, stream>>>(cbs, cnorms);

  for (int s = 0; s < RVQ_K; ++s) {
    const float* CB = cbs + (long)s * RVQ_C * RVQ_D;
    rvq_cvt<<<(RVQ_C * RVQ_D) / (256 * 8), 256, 0, stream>>>(CB, cbBf);
    rvq_argmin<<<RVQ_B / 16, 256, 0, stream>>>(residualBf, cbBf,
                                               cnorms + (long)s * RVQ_C, bestIdx);
    rvq_update<<<RVQ_B, 128, 0, stream>>>(z, CB, bestIdx, residual, residualBf,
                                          quant, outIdx, lossOut, s);
  }
}
